// MambaModel_simple_83872121356677
// MI455X (gfx1250) — matrix operand order repaired and checked by reading, not compiled, not run on hardware
//
#include <hip/hip_runtime.h>
#include <math.h>

// ---------------- model constants ----------------
#define BATCH   32
#define SEQ     360
#define DMODEL  256
#define DINNER  512
#define DSTATE  16
#define DTRANK  16
#define NBLOCKS 4
#define MROWS   (BATCH * SEQ)          // 11520

typedef __attribute__((ext_vector_type(16))) _Float16 v16h;
typedef __attribute__((ext_vector_type(8)))  _Float16 h8;
typedef __attribute__((ext_vector_type(8)))  float    v8f;

// ---------------- f32 -> f16 conversion ----------------
__global__ __launch_bounds__(256) void k_f32_to_f16(const float* __restrict__ s,
                                                    _Float16* __restrict__ d, int n) {
    int i = blockIdx.x * 256 + threadIdx.x;
    if (i < n) d[i] = (_Float16)s[i];
}

// ---------------- embedding + positional encoding ----------------
__global__ __launch_bounds__(256) void k_embed(const float* __restrict__ x,
                                               const float* __restrict__ ew,
                                               const float* __restrict__ eb,
                                               const float* __restrict__ pe,
                                               float* __restrict__ h) {
    int i = blockIdx.x * 256 + threadIdx.x;
    if (i >= MROWS * DMODEL) return;
    int d = i & (DMODEL - 1);
    int m = i >> 8;                    // DMODEL == 256
    int l = m % SEQ;
    h[i] = x[m * 2] * ew[d * 2] + x[m * 2 + 1] * ew[d * 2 + 1] + eb[d] + pe[l * DMODEL + d];
}

// ---------------- layernorm (one 256-wide row per block), f16 output ----------------
__global__ __launch_bounds__(256) void k_ln_f16(const float* __restrict__ h,
                                                const float* __restrict__ g,
                                                const float* __restrict__ b,
                                                _Float16* __restrict__ out) {
    __shared__ float red[256];
    int row = blockIdx.x, d = threadIdx.x;
    float v = h[(size_t)row * DMODEL + d];
    red[d] = v; __syncthreads();
    for (int s = 128; s > 0; s >>= 1) { if (d < s) red[d] += red[d + s]; __syncthreads(); }
    float mean = red[0] * (1.0f / DMODEL);
    __syncthreads();
    float c = v - mean;
    red[d] = c * c; __syncthreads();
    for (int s = 128; s > 0; s >>= 1) { if (d < s) red[d] += red[d + s]; __syncthreads(); }
    float rstd = rsqrtf(red[0] * (1.0f / DMODEL) + 1e-5f);
    out[(size_t)row * DMODEL + d] = (_Float16)(c * rstd * g[d] + b[d]);
}

// ---------------- WMMA GEMM:  Dst[M,NN] = A[M,KK] * Bw[NN,KK]^T (+ resid) ----------------
// NN, KK, RESID compile-time: store addresses become base + immediate offsets, no branches.
// wave32 fragment layouts used here (A/B: contiguous k, same in both; C/D as in CDNA5 ISA 7.12.2):
//   A frag : lane -> row m = lane&15 ; halves = K[k0+16*hi .. +15]   (contiguous 32B)
//   B frag : lane -> col n = lane&15 ; halves = K[k0+16*hi .. +15]   (contiguous 32B)
//   C/D    : lane -> n = lane&15 ; vgpr r -> m = r + 8*hi
template<int WM, int WN, int NN, int KK, bool RESID>
__global__ __launch_bounds__(256) void k_gemm_f16(const _Float16* __restrict__ A,
                                                  const _Float16* __restrict__ Bw,
                                                  float* __restrict__ Dst,
                                                  const float* __restrict__ resid,
                                                  int Mr) {
    const int lane = threadIdx.x & 31;
    const int wave = threadIdx.x >> 5;
    const int gw   = blockIdx.x * 8 + wave;
    const int tnCount = NN / (16 * WN);
    const int tmCount = Mr / (16 * WM);
    if (gw >= tmCount * tnCount) return;
    const int tn = gw % tnCount;
    const int tm = gw / tnCount;
    const int hi = lane >> 4;
    const int lo = lane & 15;
    const int m0 = tm * 16 * WM;
    const int n0 = tn * 16 * WN;

    v8f acc[WM][WN];
#pragma unroll
    for (int i = 0; i < WM; ++i)
#pragma unroll
        for (int j = 0; j < WN; ++j) acc[i][j] = (v8f)0.0f;

    const _Float16* ap = A  + (size_t)(m0 + lo) * KK + 16 * hi;
    const _Float16* bp = Bw + (size_t)(n0 + lo) * KK + 16 * hi;

    for (int k0 = 0; k0 < KK; k0 += 32) {
        v16h af[WM], bf[WN];
#pragma unroll
        for (int i = 0; i < WM; ++i) {
            const _Float16* p = ap + (size_t)i * 16 * KK + k0;
            h8 x0 = *(const h8*)(p);
            h8 x1 = *(const h8*)(p + 8);
            af[i] = __builtin_shufflevector(x0, x1, 0,1,2,3,4,5,6,7,8,9,10,11,12,13,14,15);
        }
#pragma unroll
        for (int j = 0; j < WN; ++j)
            bf[j] = *(const v16h*)(bp + (size_t)j * 16 * KK + k0);
#pragma unroll
        for (int i = 0; i < WM; ++i)
#pragma unroll
            for (int j = 0; j < WN; ++j)
                acc[i][j] = __builtin_amdgcn_wmma_f32_16x16x32_f16(
                    false, af[i], false, bf[j], (short)0, acc[i][j], false, false);
    }
#pragma unroll
    for (int i = 0; i < WM; ++i)
#pragma unroll
        for (int j = 0; j < WN; ++j) {
            size_t base = (size_t)(m0 + i * 16 + 8 * hi) * NN + n0 + j * 16 + lo;
            float*       dp = Dst + base;
            const float* rp = resid + base;
#pragma unroll
            for (int r = 0; r < 8; ++r) {
                float v = acc[i][j][r];
                if (RESID) v += rp[r * NN];        // compile-time gate, immediate offsets
                dp[r * NN] = v;
            }
        }
}

// ---------------- causal depthwise conv (K=4) + SiLU; emits f32 + f16 ----------------
__global__ __launch_bounds__(256) void k_conv(const float* __restrict__ xz,
                                              const float* __restrict__ cw,
                                              const float* __restrict__ cb,
                                              float* __restrict__ xcf,
                                              _Float16* __restrict__ xc16) {
    int i = blockIdx.x * 256 + threadIdx.x;
    if (i >= MROWS * DINNER) return;
    int d = i & (DINNER - 1);
    int m = i >> 9;                    // DINNER == 512
    int l = m % SEQ;
    float s = cb[d];
#pragma unroll
    for (int k = 0; k < 4; ++k) {
        int off = k - 3;
        if (l + off >= 0) s += xz[(size_t)(m + off) * (2 * DINNER) + d] * cw[d * 4 + k];
    }
    float y = s / (1.0f + __expf(-s));
    xcf[i]  = y;
    xc16[i] = (_Float16)y;
}

// ---------------- dt projection (K=16) + softplus ----------------
__global__ __launch_bounds__(256) void k_dtproj(const float* __restrict__ dbl,
                                                const float* __restrict__ dw,
                                                const float* __restrict__ db,
                                                float* __restrict__ dt) {
    int i = blockIdx.x * 256 + threadIdx.x;
    if (i >= MROWS * DINNER) return;
    int d = i & (DINNER - 1);
    int m = i >> 9;
    float s = db[d];
#pragma unroll
    for (int r = 0; r < DTRANK; ++r) s += dbl[(size_t)m * 48 + r] * dw[d * DTRANK + r];
    dt[i] = (s > 20.0f) ? s : log1pf(__expf(s));
}

// ---------------- selective scan, fused D-skip + SiLU(z) gating; f16 output ----------------
// block = 128 threads = one (batch, 128-channel chunk); state h[16] in registers.
// B/C for 8 timesteps staged per LDS fill -> 45 barrier pairs instead of 360.
#define TCH 8
__global__ __launch_bounds__(128) void k_scan(const float* __restrict__ dbl,
                                              const float* __restrict__ dt,
                                              const float* __restrict__ xcf,
                                              const float* __restrict__ xz,
                                              const float* __restrict__ A_log,
                                              const float* __restrict__ Dp,
                                              _Float16* __restrict__ y16) {
    __shared__ float sBC[TCH * 32];    // [t][0..15]=B, [t][16..31]=C
    int b = blockIdx.x >> 2;           // DINNER/128 = 4 chunks
    int d = (blockIdx.x & 3) * 128 + threadIdx.x;
    int t = threadIdx.x;
    float a[DSTATE], h[DSTATE];
#pragma unroll
    for (int n = 0; n < DSTATE; ++n) {
        a[n] = -__expf(A_log[d * DSTATE + n]);
        h[n] = 0.0f;
    }
    float Dd = Dp[d];
    for (int l0 = 0; l0 < SEQ; l0 += TCH) {
        // stage B/C for TCH steps: 128 threads x 2 loads = 256 floats
        {
            int tt = t >> 5, nn = t & 31;
            size_t r0 = (size_t)(b * SEQ + l0);
            sBC[tt * 32 + nn]            = dbl[(r0 + tt) * 48 + 16 + nn];
            sBC[(tt + 4) * 32 + nn]      = dbl[(r0 + tt + 4) * 48 + 16 + nn];
        }
        __syncthreads();
#pragma unroll
        for (int ll = 0; ll < TCH; ++ll) {
            int row = b * SEQ + l0 + ll;
            float dtv = dt [(size_t)row * DINNER + d];
            float xv  = xcf[(size_t)row * DINNER + d];
            float zv  = xz [(size_t)row * (2 * DINNER) + DINNER + d];
            const float* B = &sBC[ll * 32];
            float accv = 0.0f;
#pragma unroll
            for (int n = 0; n < DSTATE; ++n) {
                h[n] = __expf(dtv * a[n]) * h[n] + dtv * B[n] * xv;
                accv += h[n] * B[16 + n];
            }
            float y = (accv + Dd * xv) * (zv / (1.0f + __expf(-zv)));
            y16[(size_t)row * DINNER + d] = (_Float16)y;
        }
        __syncthreads();
    }
}

// ---------------- mean pool over SEQ + final fc ----------------
__global__ __launch_bounds__(256) void k_pool_fc(const float* __restrict__ h,
                                                 const float* __restrict__ fw,
                                                 const float* __restrict__ fb,
                                                 float* __restrict__ out) {
    __shared__ float pooled[DMODEL];
    int b = blockIdx.x, d = threadIdx.x;
    float s = 0.0f;
    for (int l = 0; l < SEQ; ++l) s += h[((size_t)b * SEQ + l) * DMODEL + d];
    pooled[d] = s * (1.0f / SEQ);
    __syncthreads();
    if (d < 3) {
        float o = fb[d];
        for (int j = 0; j < DMODEL; ++j) o += pooled[j] * fw[d * DMODEL + j];
        out[b * 3 + d] = o;
    }
}

// ---------------- host launch ----------------
extern "C" void kernel_launch(void* const* d_in, const int* in_sizes, int n_in,
                              void* d_out, int out_size, void* d_ws, size_t ws_size,
                              hipStream_t stream) {
    const float* x     = (const float*)d_in[0];
    const float* emb_w = (const float*)d_in[1];
    const float* emb_b = (const float*)d_in[2];
    const float* pe    = (const float*)d_in[3];
    const float* ln_g  = (const float*)d_in[4];
    const float* ln_b  = (const float*)d_in[5];
    const float* inp_w = (const float*)d_in[6];
    const float* cv_w  = (const float*)d_in[7];
    const float* cv_b  = (const float*)d_in[8];
    const float* xp_w  = (const float*)d_in[9];
    const float* dt_w  = (const float*)d_in[10];
    const float* dt_b  = (const float*)d_in[11];
    const float* A_log = (const float*)d_in[12];
    const float* Dsk   = (const float*)d_in[13];
    const float* out_w = (const float*)d_in[14];
    const float* fc_w  = (const float*)d_in[15];
    const float* fc_b  = (const float*)d_in[16];
    float* out = (float*)d_out;

    char* ws = (char*)d_ws;
    size_t off = 0;
    auto alloc = [&](size_t bytes) -> void* {
        void* p = ws + off;
        off = (off + bytes + 255) & ~(size_t)255;
        return p;
    };
    float*    hbuf   = (float*)   alloc((size_t)MROWS * DMODEL * 4);
    _Float16* hn16   = (_Float16*)alloc((size_t)MROWS * DMODEL * 2);
    _Float16* w_in16 = (_Float16*)alloc((size_t)NBLOCKS * 2 * DINNER * DMODEL * 2);
    _Float16* w_xp16 = (_Float16*)alloc((size_t)NBLOCKS * 48 * DINNER * 2);
    _Float16* w_ou16 = (_Float16*)alloc((size_t)NBLOCKS * DMODEL * DINNER * 2);
    float*    xz     = (float*)   alloc((size_t)MROWS * 2 * DINNER * 4);
    float*    xcf    = (float*)   alloc((size_t)MROWS * DINNER * 4);
    _Float16* xc16   = (_Float16*)alloc((size_t)MROWS * DINNER * 2);
    float*    dbl    = (float*)   alloc((size_t)MROWS * 48 * 4);
    float*    dtb    = (float*)   alloc((size_t)MROWS * DINNER * 4);
    _Float16* y16    = (_Float16*)alloc((size_t)MROWS * DINNER * 2);

    // Convert weights to f16 once (L2-resident afterwards).
    int n_in16 = NBLOCKS * 2 * DINNER * DMODEL;
    int n_xp16 = NBLOCKS * 48 * DINNER;
    int n_ou16 = NBLOCKS * DMODEL * DINNER;
    k_f32_to_f16<<<(n_in16 + 255) / 256, 256, 0, stream>>>(inp_w, w_in16, n_in16);
    k_f32_to_f16<<<(n_xp16 + 255) / 256, 256, 0, stream>>>(xp_w,  w_xp16, n_xp16);
    k_f32_to_f16<<<(n_ou16 + 255) / 256, 256, 0, stream>>>(out_w, w_ou16, n_ou16);

    k_embed<<<(MROWS * DMODEL + 255) / 256, 256, 0, stream>>>(x, emb_w, emb_b, pe, hbuf);

    for (int i = 0; i < NBLOCKS; ++i) {
        k_ln_f16<<<MROWS, 256, 0, stream>>>(hbuf, ln_g + i * DMODEL, ln_b + i * DMODEL, hn16);

        // in_proj: 11520 x 1024, K=256 ; waves = (11520/64)*(1024/32) = 5760
        {
            int waves = (MROWS / 64) * ((2 * DINNER) / 32);
            k_gemm_f16<4, 2, 2 * DINNER, DMODEL, false><<<(waves + 7) / 8, 256, 0, stream>>>(
                hn16, w_in16 + (size_t)i * 2 * DINNER * DMODEL, xz, nullptr, MROWS);
        }

        k_conv<<<(MROWS * DINNER + 255) / 256, 256, 0, stream>>>(
            xz, cv_w + i * DINNER * 4, cv_b + i * DINNER, xcf, xc16);

        // x_proj: 11520 x 48, K=512 ; waves = (11520/64)*(48/16) = 540
        {
            int waves = (MROWS / 64) * (48 / 16);
            k_gemm_f16<4, 1, 48, DINNER, false><<<(waves + 7) / 8, 256, 0, stream>>>(
                xc16, w_xp16 + (size_t)i * 48 * DINNER, dbl, nullptr, MROWS);
        }

        k_dtproj<<<(MROWS * DINNER + 255) / 256, 256, 0, stream>>>(
            dbl, dt_w + i * DINNER * DTRANK, dt_b + i * DINNER, dtb);

        k_scan<<<BATCH * (DINNER / 128), 128, 0, stream>>>(
            dbl, dtb, xcf, xz, A_log + (size_t)i * DINNER * DSTATE, Dsk + i * DINNER, y16);

        // out_proj (+residual): 11520 x 256, K=512 ; waves = (11520/64)*(256/32) = 1440
        {
            int waves = (MROWS / 64) * (DMODEL / 32);
            k_gemm_f16<4, 2, DMODEL, DINNER, true><<<(waves + 7) / 8, 256, 0, stream>>>(
                y16, w_ou16 + (size_t)i * DMODEL * DINNER, hbuf, hbuf, MROWS);
        }
    }

    k_pool_fc<<<BATCH, 256, 0, stream>>>(hbuf, fc_w, fc_b, out);
}
